// TemporalAttentionLayer_64716567216798
// MI455X (gfx1250) — matrix operand order repaired and checked by reading, not compiled, not run on hardware
//
#include <hip/hip_runtime.h>

// MI455X / gfx1250, wave32. bf16 WMMA with f32 accumulation.
//
// Sizes: N=16384, T=32, F=256.
// Block = 256 threads = 8 waves, 16 samples per block, grid = 1024.
// Dynamic LDS = 256KB projT (all 16 samples) + 8 waves * 6KB staging = 304KB.
// d_ws holds fragment-packed bf16 weights: W2^T frags (4MB) + W1 (128KB) + W (128KB).

typedef __attribute__((ext_vector_type(16))) __bf16 bf16x16;
typedef __attribute__((ext_vector_type(8)))  __bf16 bf16x8;
typedef __attribute__((ext_vector_type(8)))  float  f32x8;

#define NTOT 16384
#define TDIM 32
#define FDIM 256
#define NB   16
#define SMEM_BYTES (262144 + 8*6144)   // 311296

// Native f32 -> bf16 conversion; on gfx1250 this lowers to hardware cvt
// (packed v_cvt_pk_bf16_f32 when the compiler can pair lanes) instead of the
// 4-VALU-op round-to-nearest-even bit sequence.
static __device__ inline __bf16 f2bf(float f) { return (__bf16)f; }

static __device__ inline f32x8 zero8() {
  f32x8 v;
#pragma unroll
  for (int i = 0; i < 8; i++) v[i] = 0.f;
  return v;
}

static __device__ inline f32x8 wmma_bf16(bf16x16 a, bf16x16 b, f32x8 c) {
  return __builtin_amdgcn_wmma_f32_16x16x32_bf16(false, a, false, b, (short)0, c,
                                                 false, false);
}

static __device__ inline bf16x16 cat8(bf16x8 lo, bf16x8 hi) {
  return __builtin_shufflevector(lo, hi, 0,1,2,3,4,5,6,7,8,9,10,11,12,13,14,15);
}

// B-fragment from global f32: 16 consecutive floats -> e0..15 (k = 16*half + e)
static __device__ inline bf16x16 ldB16g(const float* p) {
  const float4* q = (const float4*)p;
  float4 v0 = q[0], v1 = q[1], v2 = q[2], v3 = q[3];
  bf16x16 r;
  r[0]=f2bf(v0.x); r[1]=f2bf(v0.y); r[2]=f2bf(v0.z); r[3]=f2bf(v0.w);
  r[4]=f2bf(v1.x); r[5]=f2bf(v1.y); r[6]=f2bf(v1.z); r[7]=f2bf(v1.w);
  r[8]=f2bf(v2.x); r[9]=f2bf(v2.y); r[10]=f2bf(v2.z); r[11]=f2bf(v2.w);
  r[12]=f2bf(v3.x); r[13]=f2bf(v3.y); r[14]=f2bf(v3.z); r[15]=f2bf(v3.w);
  return r;
}

// A-fragment from global f32: e0..7 = p[0..7], e8..15 = p[16..23]
static __device__ inline bf16x16 ldA16g(const float* p) {
  const float4* q = (const float4*)p;
  float4 v0 = q[0], v1 = q[1], v2 = q[4], v3 = q[5];
  bf16x16 r;
  r[0]=f2bf(v0.x); r[1]=f2bf(v0.y); r[2]=f2bf(v0.z); r[3]=f2bf(v0.w);
  r[4]=f2bf(v1.x); r[5]=f2bf(v1.y); r[6]=f2bf(v1.z); r[7]=f2bf(v1.w);
  r[8]=f2bf(v2.x); r[9]=f2bf(v2.y); r[10]=f2bf(v2.z); r[11]=f2bf(v2.w);
  r[12]=f2bf(v3.x); r[13]=f2bf(v3.y); r[14]=f2bf(v3.z); r[15]=f2bf(v3.w);
  return r;
}

// ---------------------------------------------------------------------------
// Prep: pack weights into per-lane-contiguous bf16 WMMA fragments.
//   frags 0..4095    : A-frags of W2[s]^T  (A[m=f][k=g] = W2[s][g][f])
//   frags 4096..4223 : B-frags of W1       (B[k=g][n=f] = W1[g][f])
//   frags 4224..4351 : B-frags of W        (B[k=g][n=f] = W [g][f])
// Fragment = 32 lanes x 16 bf16 = 1024 bytes.
// ---------------------------------------------------------------------------
__global__ __launch_bounds__(256) void pack_weights_kernel(
    const float* __restrict__ W, const float* __restrict__ W1,
    const float* __restrict__ W2, __bf16* __restrict__ ws) {
  const int fi = blockIdx.x;
  const int tid = threadIdx.x;
#pragma unroll
  for (int half = 0; half < 2; half++) {
    const int idx = tid + half * 256;       // 0..511 within fragment
    const int lane = idx >> 4, e = idx & 15;
    const int lh = lane >> 4, lm = lane & 15;
    float v;
    if (fi < 4096) {                        // W2^T A-fragments
      const int s = fi >> 7, rem = fi & 127;
      const int mtf = rem >> 3, ktg = rem & 7;
      const int f = mtf * 16 + lm;
      const int g = ktg * 32 + lh * 16 + e;
      v = W2[((size_t)s * FDIM + g) * FDIM + f];
    } else if (fi < 4224) {                 // W1 B-fragments
      const int r = fi - 4096, ntf = r >> 3, ktg = r & 7;
      const int f = ntf * 16 + lm;
      const int g = ktg * 32 + ((e < 8) ? (lh * 8 + e) : (16 + lh * 8 + (e - 8)));
      v = W1[(size_t)g * FDIM + f];
    } else {                                // W B-fragments
      const int r = fi - 4224, ntf = r >> 3, ktg = r & 7;
      const int f = ntf * 16 + lm;
      const int g = ktg * 32 + ((e < 8) ? (lh * 8 + e) : (16 + lh * 8 + (e - 8)));
      v = W[(size_t)g * FDIM + f];
    }
    ws[(size_t)fi * 512 + idx] = f2bf(v);
  }
}

// ---------------------------------------------------------------------------
// Main fused kernel.
// ---------------------------------------------------------------------------
__global__ __launch_bounds__(256) void temporal_attention_kernel(
    const float* __restrict__ X, const __bf16* __restrict__ wsb,
    float* __restrict__ out) {
  extern __shared__ char smem[];
  const __bf16* w2t = wsb;                      // 4096 frags
  const __bf16* w1b = wsb + (size_t)4096 * 512; // 128 frags
  const __bf16* wbt = wsb + (size_t)4224 * 512; // 128 frags

  const int tid = threadIdx.x;
  const int wave = tid >> 5, lane = tid & 31;
  const int lh = lane >> 4, lm = lane & 15;
  const int n0 = blockIdx.x * NB;

  char* projT = smem;                            // [n<16][f<256][s<32] bf16
  char* wsc  = smem + 262144 + wave * 6144;      // per-wave staging
  char* h1s  = wsc;                              // [t<16][f<32]  bf16 (64B rows)
  char* h2s  = wsc + 1024;                       // [st<2][s'<16][f<32]
  char* hts  = wsc + 3072;                       // [f<16][s<32]
  char* alph = wsc + 4096;                       // [t<32][s<32]

  // ---- Phase P (cooperative): projT[n][f][s] = (X[n,s,:] @ W2[s])[f] ------
  // Flipped GEMM per s: M = f (16 tiles), N = n (16 samples), K = g (8x32).
  for (int i = 0; i < 4; i++) {
    const int s = wave * 4 + i;
    bf16x16 xb[8];
    const float* xp = X + (((size_t)(n0 + lm)) * TDIM + s) * FDIM + lh * 16;
#pragma unroll
    for (int kt = 0; kt < 8; kt++) xb[kt] = ldB16g(xp + kt * 32);
    if (i < 3)
      __builtin_prefetch(X + (((size_t)(n0 + lm)) * TDIM + s + 1) * FDIM, 0, 3);
    const __bf16* wfs = w2t + (size_t)(s * 16) * 8 * 512 + (size_t)lane * 16;
    for (int mtf = 0; mtf < 16; mtf++) {
      f32x8 acc = zero8();
      const __bf16* wf = wfs + (size_t)mtf * 8 * 512;
#pragma unroll
      for (int kt = 0; kt < 8; kt++)
        acc = wmma_bf16(*(const bf16x16*)(wf + kt * 512), xb[kt], acc);
      // C tile: row m = f, col n = sample(lane%16); scatter to projT[n][f][s]
      char* base = projT + lm * 16384 + s * 2;
#pragma unroll
      for (int g = 0; g < 8; g++)
        *(__bf16*)(base + (mtf * 16 + lh * 8 + g) * 64) = f2bf(acc[g]);
    }
  }
  __syncthreads();

  // ---- Per-wave phases: 2 samples per wave, fully wave-local --------------
  for (int sm = 0; sm < 2; sm++) {
    const int nl = wave * 2 + sm;
    const size_t nab = (size_t)n0 + nl;
    const float* Xs = X + nab * TDIM * FDIM;

    // Cache X A-fragments for this sample's 32 rows (rows = t and rows = s).
    bf16x16 xa[2][8];
#pragma unroll
    for (int mt = 0; mt < 2; mt++)
#pragma unroll
      for (int kt = 0; kt < 8; kt++)
        xa[mt][kt] = ldA16g(Xs + (size_t)(mt * 16 + lm) * FDIM + kt * 32 + lh * 8);

    // Inline constant L A-fragments: L[t,s] = (s<=t)/(t+1)  (cumsum+denominator)
    bf16x16 la[2];
#pragma unroll
    for (int st = 0; st < 2; st++) {
      const int trow = st * 16 + lm;
#pragma unroll
      for (int e = 0; e < 16; e++) {
        const int sidx = lh * 16 + e;
        la[st][e] = f2bf(sidx <= trow ? 1.0f / (float)(trow + 1) : 0.0f);
      }
    }

    // ---- alph = (X@W1) @ (L@proj)^T, fused per 32-wide f-chunk ----
    f32x8 accA[2][2];
#pragma unroll
    for (int a = 0; a < 2; a++)
#pragma unroll
      for (int b = 0; b < 2; b++) accA[a][b] = zero8();

    for (int kt = 0; kt < 8; kt++) {
      // H2 chunk [16 s'][32 f] per st: single K=32 WMMA over s, A = L.
#pragma unroll
      for (int st = 0; st < 2; st++) {
#pragma unroll
        for (int fs = 0; fs < 2; fs++) {
          bf16x16 pb = *(const bf16x16*)(projT + nl * 16384 +
                                         (kt * 32 + fs * 16 + lm) * 64 + lh * 32);
          f32x8 c = wmma_bf16(la[st], pb, zero8());
          char* hb = h2s + st * 1024;
#pragma unroll
          for (int g = 0; g < 8; g++)
            *(__bf16*)(hb + (lh * 8 + g) * 64 + (fs * 16 + lm) * 2) = f2bf(c[g]);
        }
      }
#pragma unroll
      for (int mt = 0; mt < 2; mt++) {
        // H1 chunk [16 t][32 f] = X @ W1 (8 K-steps over g), staged to LDS.
#pragma unroll
        for (int fs = 0; fs < 2; fs++) {
          f32x8 c = zero8();
          const __bf16* wf = w1b + (size_t)((kt * 2 + fs) * 8) * 512 + lane * 16;
#pragma unroll
          for (int kg = 0; kg < 8; kg++)
            c = wmma_bf16(xa[mt][kg], *(const bf16x16*)(wf + kg * 512), c);
#pragma unroll
          for (int g = 0; g < 8; g++)
            *(__bf16*)(h1s + (lh * 8 + g) * 64 + (fs * 16 + lm) * 2) = f2bf(c[g]);
        }
        // Re-read as A-fragment (rows t, k = f).
        bf16x16 af = cat8(*(const bf16x8*)(h1s + lm * 64 + lh * 32),
                          *(const bf16x8*)(h1s + lm * 64 + lh * 32 + 16));
#pragma unroll
        for (int st = 0; st < 2; st++)
          accA[mt][st] = wmma_bf16(
              af, *(const bf16x16*)(h2s + st * 1024 + lm * 64 + lh * 32),
              accA[mt][st]);
      }
    }

    // alph -> LDS [t][s] bf16
#pragma unroll
    for (int mt = 0; mt < 2; mt++)
#pragma unroll
      for (int st = 0; st < 2; st++)
#pragma unroll
        for (int g = 0; g < 8; g++)
          *(__bf16*)(alph + (mt * 16 + lh * 8 + g) * 64 + (st * 16 + lm) * 2) =
              f2bf(accA[mt][st][g]);

    bf16x16 aa[2];
#pragma unroll
    for (int mt = 0; mt < 2; mt++)
      aa[mt] = cat8(*(const bf16x8*)(alph + (mt * 16 + lm) * 64 + lh * 32),
                    *(const bf16x8*)(alph + (mt * 16 + lm) * 64 + lh * 32 + 16));

    // ---- out = alph @ (X@W) + X, fused per 16-wide f-chunk ----
    for (int nt = 0; nt < 16; nt++) {
      // Ht chunk [32 s][16 f], stored transposed [f][s] with packed 16B stores.
#pragma unroll
      for (int m2 = 0; m2 < 2; m2++) {
        f32x8 c = zero8();
        const __bf16* wf = wbt + (size_t)(nt * 8) * 512 + lane * 16;
#pragma unroll
        for (int kg = 0; kg < 8; kg++)
          c = wmma_bf16(xa[m2][kg], *(const bf16x16*)(wf + kg * 512), c);
        bf16x8 pk;
#pragma unroll
        for (int g = 0; g < 8; g++) pk[g] = f2bf(c[g]);
        *(bf16x8*)(hts + lm * 64 + (m2 * 16 + lh * 8) * 2) = pk;
      }
#pragma unroll
      for (int mt = 0; mt < 2; mt++) {
        bf16x16 hb = *(const bf16x16*)(hts + lm * 64 + lh * 32);  // B: n=f, k=s
        f32x8 c = wmma_bf16(aa[mt], hb, zero8());                  // K = s = 32
        const size_t rowb =
            (nab * TDIM + mt * 16 + lh * 8) * FDIM + nt * 16 + lm;
#pragma unroll
        for (int g = 0; g < 8; g++) {
          const size_t off = rowb + (size_t)g * FDIM;
          out[off] = c[g] + X[off];   // residual
        }
      }
    }
  }
}

extern "C" void kernel_launch(void* const* d_in, const int* in_sizes, int n_in,
                              void* d_out, int out_size, void* d_ws, size_t ws_size,
                              hipStream_t stream) {
  (void)in_sizes; (void)n_in; (void)out_size; (void)ws_size;
  const float* X  = (const float*)d_in[0];   // inputs        [N,T,F]
  const float* W  = (const float*)d_in[1];   // temp_weights  [F,F]
  const float* W1 = (const float*)d_in[2];   // temp1_weights [F,F]
  const float* W2 = (const float*)d_in[3];   // temp2_weights [T,F,F]
  float* out = (float*)d_out;
  __bf16* wsb = (__bf16*)d_ws;               // needs ~4.25 MB

  hipFuncSetAttribute(reinterpret_cast<const void*>(temporal_attention_kernel),
                      hipFuncAttributeMaxDynamicSharedMemorySize, SMEM_BYTES);

  pack_weights_kernel<<<4352, 256, 0, stream>>>(W, W1, W2, wsb);
  temporal_attention_kernel<<<NTOT / NB, 256, SMEM_BYTES, stream>>>(X, wsb, out);
}
